// GraphMambaLayer_14972255994597
// MI455X (gfx1250) — matrix operand order repaired and checked by reading, not compiled, not run on hardware
//
#include <hip/hip_runtime.h>
#include <math.h>
#include <stdint.h>

// ---------------- CDNA5 WMMA / TDM types ----------------
typedef __attribute__((ext_vector_type(16))) __bf16 v16bf;
typedef __attribute__((ext_vector_type(8)))  float  v8f;
typedef __attribute__((ext_vector_type(4)))  unsigned int u32x4;
typedef __attribute__((ext_vector_type(8)))  int i32x8;
typedef __attribute__((ext_vector_type(4)))  int i32x4;

union Frag {
    v16bf v;
    uint4 q[2];
};

// Problem constants (from reference)
#define NROW 16384        // B*K
#define DD   512
#define NLAY 2

#define BM 128
#define BN 128
#define BK 32

// ---------------- TDM: 2D tile load, bf16 (data_size=2B) ----------------
// D# packing per CDNA5 ISA:
//  group0: [1:0]=count=1, [63:32]=lds_addr, [120:64]=global_addr, [127:126]=type=2
//  group1: [17:16]=data_size(1->2B), [79:48]=tensor_dim0, [111:80]=tensor_dim1,
//          [127:112]=tile_dim0, [143:128]=tile_dim1, [207:160]=tensor_dim0_stride
// Toolchain here uses the 6-arg builtin form:
//  (u32x4 g0, i32x8 g1, i32x4 g2, i32x4 g3, i32x8 g4, i32 cpol)
__device__ __forceinline__ void tdm_load_2d(unsigned lds_off, const void* gptr,
                                            unsigned tile0, unsigned tile1,
                                            unsigned long long stride0)
{
    const unsigned long long ga = (unsigned long long)(uintptr_t)gptr;
    const unsigned tdim0 = tile0, tdim1 = tile1;   // tensor dims >= tile dims (no clip)
    u32x4 g0;
    g0[0] = 1u;                                            // count=1, user mode
    g0[1] = lds_off;                                       // LDS byte address
    g0[2] = (unsigned)ga;                                  // global_addr[31:0]
    g0[3] = (unsigned)((ga >> 32) & 0x1FFFFFFu) | (2u << 30); // addr[56:32] | type=2
    i32x8 g1;
    g1[0] = (int)(1u << 16);                               // data_size=1 (2 bytes)
    g1[1] = (int)((tdim0 & 0xFFFFu) << 16);                // tensor_dim0[15:0]
    g1[2] = (int)(((tdim0 >> 16) & 0xFFFFu) | ((tdim1 & 0xFFFFu) << 16));
    g1[3] = (int)(((tdim1 >> 16) & 0xFFFFu) | (tile0 << 16)); // tile_dim0
    g1[4] = (int)tile1;                                    // tile_dim1 (tile_dim2=0)
    g1[5] = (int)(unsigned)(stride0 & 0xFFFFFFFFu);        // dim0 stride (elements)
    g1[6] = (int)(unsigned)((stride0 >> 32) & 0xFFFFu);
    g1[7] = 0;
    const i32x4 gz4 = {0, 0, 0, 0};
    const i32x8 gz8 = {0, 0, 0, 0, 0, 0, 0, 0};
    __builtin_amdgcn_tensor_load_to_lds(g0, g1, gz4, gz4, gz8, 0);
}

__device__ __forceinline__ float gelu_f(float v)
{
    // branch-free tanh-form gelu
    const float u = v * (0.7978845608f + 0.0356774081f * v * v);
    const float e = __expf(2.0f * u);
    return 0.5f * v * (1.0f + (e - 1.0f) / (e + 1.0f));
}

// ---------------- Tiled bf16 WMMA GEMM, TDM-staged, double-buffered ----------------
// C[16384, Ncols] = A(bf16,lda) @ W(bf16,ldw) + bias, epilogue EPI.
// EPI: 0 bf16 | 1 gelu->bf16 | 2 (+res)->f32 | 3 (+res)->bf16 | 4 (+res)*mask->f32
template <int EPI>
__global__ __launch_bounds__(256) void gemm_bf16_k(
    const __bf16* __restrict__ A, int lda,
    const __bf16* __restrict__ W, int ldw,
    const float*  __restrict__ bias,
    const float*  __restrict__ res, int ldr,
    const float*  __restrict__ mask,
    void* __restrict__ outv, int ldo,
    int Kdim)
{
    __shared__ alignas(16) __bf16 As[2][BM][BK];
    __shared__ alignas(16) __bf16 Bs[2][BK][BN];

    const int tid  = threadIdx.x;
    const int wid  = tid >> 5;
    const int lane = tid & 31;
    const int wrow = wid >> 2;        // 0..1  (64-row band)
    const int wcol = wid & 3;         // 0..3  (32-col band)
    const int lhi  = lane >> 4;       // 0/1
    const int llo  = lane & 15;
    const int m0   = blockIdx.y * BM;
    const int n0   = blockIdx.x * BN;
    const int nk   = Kdim / BK;

    v8f acc[4][2];
#pragma unroll
    for (int mi = 0; mi < 4; ++mi)
#pragma unroll
        for (int ni = 0; ni < 2; ++ni)
            acc[mi][ni] = v8f{0.f, 0.f, 0.f, 0.f, 0.f, 0.f, 0.f, 0.f};

    // Tensor Data Mover: wave 0 DMAs both tiles of k-slice kb into LDS buffer buf.
    auto issue_tiles = [&](int kb, int buf) {
        const __bf16* ap = A + (size_t)m0 * lda + (size_t)kb * BK;
        const __bf16* wp = W + (size_t)kb * BK * ldw + n0;
        // A tile: BM rows x BK cols, row stride lda  -> As[buf] row-major [128][32]
        tdm_load_2d((unsigned)(uintptr_t)&As[buf][0][0], ap, BK, BM,
                    (unsigned long long)lda);
        // W tile: BK rows x BN cols, row stride ldw  -> Bs[buf] row-major [32][128]
        tdm_load_2d((unsigned)(uintptr_t)&Bs[buf][0][0], wp, BN, BK,
                    (unsigned long long)ldw);
    };

    if (wid == 0) {
        issue_tiles(0, 0);
        __builtin_amdgcn_s_wait_tensorcnt(0);
    }
    __syncthreads();

    for (int kb = 0; kb < nk; ++kb) {
        const int buf = kb & 1;
        const bool more = (kb + 1) < nk;
        if (wid == 0 && more) issue_tiles(kb + 1, buf ^ 1);   // overlap DMA w/ math

        // A fragments: lane L holds M = base + (L&15); K = (v/4)*16 + (L/16)*8 + (v%4)*2 + h
        Frag af[4];
#pragma unroll
        for (int mi = 0; mi < 4; ++mi) {
            const int M = wrow * 64 + mi * 16 + llo;
            af[mi].q[0] = *reinterpret_cast<const uint4*>(&As[buf][M][lhi * 8]);
            af[mi].q[1] = *reinterpret_cast<const uint4*>(&As[buf][M][16 + lhi * 8]);
        }
        // B fragments: lane L holds N = nb + (L&15); element s -> same K order as A
        Frag bfr[2];
#pragma unroll
        for (int ni = 0; ni < 2; ++ni) {
            const int nb = wcol * 32 + ni * 16;
#pragma unroll
            for (int s = 0; s < 8; ++s) {
                bfr[ni].v[s]     = Bs[buf][lhi * 8 + s][nb + llo];
                bfr[ni].v[8 + s] = Bs[buf][16 + lhi * 8 + s][nb + llo];
            }
        }
#pragma unroll
        for (int mi = 0; mi < 4; ++mi)
#pragma unroll
            for (int ni = 0; ni < 2; ++ni)
                acc[mi][ni] = __builtin_amdgcn_wmma_f32_16x16x32_bf16(
                    false, af[mi].v, false, bfr[ni].v,
                    (short)0, acc[mi][ni], false, false);

        if (wid == 0 && more) __builtin_amdgcn_s_wait_tensorcnt(0);
        __syncthreads();
    }

    // epilogue: VGPR r -> M = base + (lane/16)*8 + r ; lanes 0..15 -> N
#pragma unroll
    for (int mi = 0; mi < 4; ++mi)
#pragma unroll
        for (int ni = 0; ni < 2; ++ni) {
            const int col   = n0 + wcol * 32 + ni * 16 + llo;
            const int rbase = m0 + wrow * 64 + mi * 16 + lhi * 8;
            const float bv  = bias[col];
#pragma unroll
            for (int r = 0; r < 8; ++r) {
                float v = acc[mi][ni][r] + bv;
                const int rr = rbase + r;
                if (EPI == 0) {
                    ((__bf16*)outv)[(size_t)rr * ldo + col] = (__bf16)v;
                } else if (EPI == 1) {
                    ((__bf16*)outv)[(size_t)rr * ldo + col] = (__bf16)gelu_f(v);
                } else if (EPI == 2) {
                    ((float*)outv)[(size_t)rr * ldo + col] =
                        v + res[(size_t)rr * ldr + col];
                } else if (EPI == 3) {
                    ((__bf16*)outv)[(size_t)rr * ldo + col] =
                        (__bf16)(v + res[(size_t)rr * ldr + col]);
                } else { // 4
                    ((float*)outv)[(size_t)rr * ldo + col] =
                        (v + res[(size_t)rr * ldr + col]) * mask[rr];
                }
            }
        }
}

// ---------------- weight f32 -> bf16 (one-time, per call) ----------------
__global__ __launch_bounds__(256) void cvt_bf16_k(const float* __restrict__ x,
                                                  __bf16* __restrict__ o)
{
    const size_t i = ((size_t)blockIdx.x * 256 + threadIdx.x) * 4;
    const float4 v = *reinterpret_cast<const float4*>(x + i);
    o[i + 0] = (__bf16)v.x;
    o[i + 1] = (__bf16)v.y;
    o[i + 2] = (__bf16)v.z;
    o[i + 3] = (__bf16)v.w;
}

// ---------------- centroids: bandwidth reduction over 4096 ----------------
__global__ __launch_bounds__(256) void centroid_k(const float* __restrict__ attn,
                                                  float* __restrict__ cent,
                                                  float* __restrict__ out_tail)
{
    const int row = blockIdx.x;                 // b*K + k
    const float* ar = attn + (size_t)row * 4096;
    float s0 = 0.f, s1 = 0.f, s2 = 0.f;
    for (int p = threadIdx.x; p < 4096; p += 256) {
        const float a  = ar[p];
        const float gx = (float)(p & 63) * (1.0f / 63.0f);
        const float gy = (float)(p >> 6) * (1.0f / 63.0f);
        s0 += a; s1 += a * gx; s2 += a * gy;
    }
    __shared__ float r0[256], r1[256], r2[256];
    r0[threadIdx.x] = s0; r1[threadIdx.x] = s1; r2[threadIdx.x] = s2;
    for (int st = 128; st > 0; st >>= 1) {
        __syncthreads();
        if (threadIdx.x < st) {
            r0[threadIdx.x] += r0[threadIdx.x + st];
            r1[threadIdx.x] += r1[threadIdx.x + st];
            r2[threadIdx.x] += r2[threadIdx.x + st];
        }
    }
    if (threadIdx.x == 0) {
        const float inv = 1.0f / (r0[0] + 1e-8f);
        const float cx = r1[0] * inv, cy = r2[0] * inv;
        cent[(size_t)row * 2 + 0] = cx;
        cent[(size_t)row * 2 + 1] = cy;
        out_tail[(size_t)row * 2 + 0] = cx;
        out_tail[(size_t)row * 2 + 1] = cy;
    }
}

// ---------------- sin/cos positional encoder -> bf16 (N x 256) ----------------
__global__ __launch_bounds__(256) void posenc_k(const float* __restrict__ cent,
                                                __bf16* __restrict__ enc)
{
    const int row = blockIdx.x;
    const int t = threadIdx.x;
    const float cx = cent[(size_t)row * 2 + 0];
    const float cy = cent[(size_t)row * 2 + 1];
    const int c = t >> 6, j = t & 63;
    const float feat = (c == 0) ? cx : (c == 1) ? cy : 0.1f;
    const int jj = j & 31;
    const float freq = __expf(-9.210340372f * ((float)jj * (1.0f / 32.0f))); // 10000^-(jj/32)
    const float ang = feat * freq * 3.14159f;
    const float val = (j < 32) ? __sinf(ang) : __cosf(ang);
    enc[(size_t)row * 256 + t] = (__bf16)val;
}

// ---------------- LayerNorm, dual affine (fwd/bwd share stats) ----------------
__global__ __launch_bounds__(256) void ln_dual_k(const float* __restrict__ x,
                                                 const float* __restrict__ g1, const float* __restrict__ b1,
                                                 const float* __restrict__ g2, const float* __restrict__ b2,
                                                 __bf16* __restrict__ o1, __bf16* __restrict__ o2)
{
    const int wid = threadIdx.x >> 5, lane = threadIdx.x & 31;
    const int row = blockIdx.x * 8 + wid;
    const float* xr = x + (size_t)row * DD;
    float v[16];
    float s = 0.f, sq = 0.f;
#pragma unroll
    for (int j = 0; j < 16; ++j) {
        const float t = xr[lane + j * 32];
        v[j] = t; s += t; sq += t * t;
    }
    for (int o = 16; o >= 1; o >>= 1) { s += __shfl_xor(s, o, 32); sq += __shfl_xor(sq, o, 32); }
    const float mean = s * (1.0f / DD);
    const float rs = rsqrtf(sq * (1.0f / DD) - mean * mean + 1e-5f);
#pragma unroll
    for (int j = 0; j < 16; ++j) {
        const int c = lane + j * 32;
        const float xh = (v[j] - mean) * rs;
        o1[(size_t)row * DD + c] = (__bf16)(xh * g1[c] + b1[c]);
        o2[(size_t)row * DD + c] = (__bf16)(xh * g2[c] + b2[c]);
    }
}

// ---------------- fused: slots = LN(y; norm) [f32] ; out = LN(slots; ffn_ln) [bf16] ----------------
__global__ __launch_bounds__(256) void ln_norm_ffn_k(const float* __restrict__ y,
                                                     const float* __restrict__ ng, const float* __restrict__ nb,
                                                     const float* __restrict__ fg, const float* __restrict__ fb,
                                                     float* __restrict__ slots, __bf16* __restrict__ o)
{
    const int wid = threadIdx.x >> 5, lane = threadIdx.x & 31;
    const int row = blockIdx.x * 8 + wid;
    const float* yr = y + (size_t)row * DD;
    float v[16];
    float s = 0.f, sq = 0.f;
#pragma unroll
    for (int j = 0; j < 16; ++j) {
        const float t = yr[lane + j * 32];
        v[j] = t; s += t; sq += t * t;
    }
    for (int oo = 16; oo >= 1; oo >>= 1) { s += __shfl_xor(s, oo, 32); sq += __shfl_xor(sq, oo, 32); }
    float mean = s * (1.0f / DD);
    float rs = rsqrtf(sq * (1.0f / DD) - mean * mean + 1e-5f);
    float sv[16];
    float s2 = 0.f, sq2 = 0.f;
#pragma unroll
    for (int j = 0; j < 16; ++j) {
        const int c = lane + j * 32;
        const float sj = (v[j] - mean) * rs * ng[c] + nb[c];
        sv[j] = sj;
        slots[(size_t)row * DD + c] = sj;
        s2 += sj; sq2 += sj * sj;
    }
    for (int oo = 16; oo >= 1; oo >>= 1) { s2 += __shfl_xor(s2, oo, 32); sq2 += __shfl_xor(sq2, oo, 32); }
    mean = s2 * (1.0f / DD);
    rs = rsqrtf(sq2 * (1.0f / DD) - mean * mean + 1e-5f);
#pragma unroll
    for (int j = 0; j < 16; ++j) {
        const int c = lane + j * 32;
        o[(size_t)row * DD + c] = (__bf16)((sv[j] - mean) * rs * fg[c] + fb[c]);
    }
}

// ---------------- GLU: act = silu(h[:, :1024]) * sigmoid(h[:, 1024:]) ----------------
__global__ __launch_bounds__(256) void glu_k(const __bf16* __restrict__ h,
                                             __bf16* __restrict__ act)
{
    const size_t e = ((size_t)blockIdx.x * 256 + threadIdx.x) * 4;
    const size_t row = e >> 10;
    const int c = (int)(e & 1023);
    const __bf16* hr = h + row * 2048;
#pragma unroll
    for (int i = 0; i < 4; ++i) {
        const float z = (float)hr[c + i];
        const float g = (float)hr[1024 + c + i];
        const float sz = 1.0f / (1.0f + __expf(-z));
        const float sg = 1.0f / (1.0f + __expf(-g));
        act[row * 1024 + c + i] = (__bf16)(z * sz * sg);
    }
}

// ---------------- host launcher ----------------
extern "C" void kernel_launch(void* const* d_in, const int* in_sizes, int n_in,
                              void* d_out, int out_size, void* d_ws, size_t ws_size,
                              hipStream_t stream)
{
    (void)in_sizes; (void)n_in; (void)out_size; (void)ws_size;
    const float* slots_in = (const float*)d_in[0];
    const float* keep     = (const float*)d_in[1];
    const float* attn     = (const float*)d_in[2];
    const float* pm1_w = (const float*)d_in[5];
    const float* pm1_b = (const float*)d_in[6];
    const float* pm2_w = (const float*)d_in[7];
    const float* pm2_b = (const float*)d_in[8];
    const float* fwd_ln_g = (const float*)d_in[9];
    const float* fwd_ln_b = (const float*)d_in[10];
    const float* fwd_in_w = (const float*)d_in[11];
    const float* fwd_in_b = (const float*)d_in[12];
    const float* fwd_out_w = (const float*)d_in[13];
    const float* fwd_out_b = (const float*)d_in[14];
    const float* bwd_ln_g = (const float*)d_in[15];
    const float* bwd_ln_b = (const float*)d_in[16];
    const float* bwd_in_w = (const float*)d_in[17];
    const float* bwd_in_b = (const float*)d_in[18];
    const float* bwd_out_w = (const float*)d_in[19];
    const float* bwd_out_b = (const float*)d_in[20];
    const float* merge_w = (const float*)d_in[21];
    const float* merge_b = (const float*)d_in[22];
    const float* norm_g = (const float*)d_in[23];
    const float* norm_b = (const float*)d_in[24];
    const float* ffn_ln_g = (const float*)d_in[25];
    const float* ffn_ln_b = (const float*)d_in[26];
    const float* ffn1_w = (const float*)d_in[27];
    const float* ffn1_b = (const float*)d_in[28];
    const float* ffn2_w = (const float*)d_in[29];
    const float* ffn2_b = (const float*)d_in[30];

    const size_t N = NROW;
    char* ws = (char*)d_ws;
    size_t off = 0;
    float*  centW = (float*)(ws + off);  off += N * 2 * sizeof(float);
    float*  xf    = (float*)(ws + off);  off += N * DD * sizeof(float);
    float*  ybuf  = (float*)(ws + off);  off += N * DD * sizeof(float);
    __bf16* xlnA  = (__bf16*)(ws + off); off += N * DD * sizeof(__bf16);
    __bf16* xlnB  = (__bf16*)(ws + off); off += N * DD * sizeof(__bf16);
    __bf16* hbuf  = (__bf16*)(ws + off); off += N * 2048 * sizeof(__bf16);
    __bf16* act   = (__bf16*)(ws + off); off += N * 1024 * sizeof(__bf16);
    __bf16* fb    = (__bf16*)(ws + off); off += N * 1024 * sizeof(__bf16);
    __bf16* enc = hbuf;     // alias: N x 256
    __bf16* g1  = act;      // alias: N x 512

    // converted bf16 weights
    auto bump = [&](size_t elems) { __bf16* p = (__bf16*)(ws + off); off += elems * sizeof(__bf16); return p; };
    __bf16* pm1c = bump(256 * 512);
    __bf16* pm2c = bump(512 * 512);
    __bf16* finc = bump((size_t)NLAY * 512 * 2048);
    __bf16* foutc = bump((size_t)NLAY * 1024 * 512);
    __bf16* binc = bump((size_t)NLAY * 512 * 2048);
    __bf16* boutc = bump((size_t)NLAY * 1024 * 512);
    __bf16* mrgc = bump((size_t)NLAY * 1024 * 512);
    __bf16* ff1c = bump((size_t)NLAY * 512 * 2048);
    __bf16* ff2c = bump((size_t)NLAY * 2048 * 512);

    // one-time (per call) weight conversion f32 -> bf16
    auto cvt = [&](const float* src, __bf16* dst, size_t elems) {
        cvt_bf16_k<<<(int)(elems / 1024), 256, 0, stream>>>(src, dst);
    };
    cvt(pm1_w, pm1c, 256 * 512);
    cvt(pm2_w, pm2c, 512 * 512);
    cvt(fwd_in_w, finc, (size_t)NLAY * 512 * 2048);
    cvt(fwd_out_w, foutc, (size_t)NLAY * 1024 * 512);
    cvt(bwd_in_w, binc, (size_t)NLAY * 512 * 2048);
    cvt(bwd_out_w, boutc, (size_t)NLAY * 1024 * 512);
    cvt(merge_w, mrgc, (size_t)NLAY * 1024 * 512);
    cvt(ffn1_w, ff1c, (size_t)NLAY * 512 * 2048);
    cvt(ffn2_w, ff2c, (size_t)NLAY * 2048 * 512);

    float* outSlots = (float*)d_out;
    float* outCent  = outSlots + N * DD;

    const dim3 g512(512 / BN, (int)(N / BM));    // (4, 128)
    const dim3 g2048(2048 / BN, (int)(N / BM));  // (16, 128)

    // 1. centroids (+ output tail)
    centroid_k<<<(int)N, 256, 0, stream>>>(attn, centW, outCent);
    // 2. sin/cos encoding -> bf16
    posenc_k<<<(int)N, 256, 0, stream>>>(centW, enc);
    // 3. pos_enc MLP
    gemm_bf16_k<1><<<g512, 256, 0, stream>>>(enc, 256, pm1c, 512, pm1_b,
                                             nullptr, 0, nullptr, (void*)g1, 512, 256);
    gemm_bf16_k<2><<<g512, 256, 0, stream>>>(g1, 512, pm2c, 512, pm2_b,
                                             slots_in, 512, nullptr, (void*)xf, 512, 512);

    for (int i = 0; i < NLAY; ++i) {
        const size_t o512 = (size_t)i * 512;
        const size_t o2048 = (size_t)i * 2048;
        const size_t oW1 = (size_t)i * 512 * 2048;
        const size_t oW2 = (size_t)i * 1024 * 512;
        const size_t oF2 = (size_t)i * 2048 * 512;

        ln_dual_k<<<(int)(N / 8), 256, 0, stream>>>(xf,
            fwd_ln_g + o512, fwd_ln_b + o512, bwd_ln_g + o512, bwd_ln_b + o512,
            xlnA, xlnB);

        // forward branch -> fb[:, :512]
        gemm_bf16_k<0><<<g2048, 256, 0, stream>>>(xlnA, 512, finc + oW1, 2048,
            fwd_in_b + o2048, nullptr, 0, nullptr, (void*)hbuf, 2048, 512);
        glu_k<<<(int)N, 256, 0, stream>>>(hbuf, act);
        gemm_bf16_k<3><<<g512, 256, 0, stream>>>(act, 1024, foutc + oW2, 512,
            fwd_out_b + o512, xf, 512, nullptr, (void*)fb, 1024, 1024);

        // backward branch (Hilbert permutation cancels) -> fb[:, 512:]
        gemm_bf16_k<0><<<g2048, 256, 0, stream>>>(xlnB, 512, binc + oW1, 2048,
            bwd_in_b + o2048, nullptr, 0, nullptr, (void*)hbuf, 2048, 512);
        glu_k<<<(int)N, 256, 0, stream>>>(hbuf, act);
        gemm_bf16_k<3><<<g512, 256, 0, stream>>>(act, 1024, boutc + oW2, 512,
            bwd_out_b + o512, xf, 512, nullptr, (void*)(fb + 512), 1024, 1024);

        // merge: y = xf + [f,bk] @ merge_w + merge_b  (f32)
        gemm_bf16_k<2><<<g512, 256, 0, stream>>>(fb, 1024, mrgc + oW2, 512,
            merge_b + o512, xf, 512, nullptr, (void*)ybuf, 512, 1024);

        // slots = LN(y; norm) ; xlnA = LN(slots; ffn_ln)
        ln_norm_ffn_k<<<(int)(N / 8), 256, 0, stream>>>(ybuf,
            norm_g + o512, norm_b + o512, ffn_ln_g + o512, ffn_ln_b + o512,
            xf, xlnA);

        // FFN
        gemm_bf16_k<1><<<g2048, 256, 0, stream>>>(xlnA, 512, ff1c + oW1, 2048,
            ffn1_b + o2048, nullptr, 0, nullptr, (void*)hbuf, 2048, 512);
        float* outp = (i == NLAY - 1) ? outSlots : xf;
        gemm_bf16_k<4><<<g512, 256, 0, stream>>>(hbuf, 2048, ff2c + oF2, 512,
            ffn2_b + o512, xf, 512, keep, (void*)outp, 512, 2048);
    }
}
